// CTC2DHead_11965778886989
// MI455X (gfx1250) — matrix operand order repaired and checked by reading, not compiled, not run on hardware
//
#include <hip/hip_runtime.h>
#include <cmath>

// ---------- types ----------
typedef __attribute__((ext_vector_type(16))) __bf16 bf16x16;
typedef __attribute__((ext_vector_type(8)))  float  f32x8;
typedef __attribute__((ext_vector_type(4)))  float  f32x4;
typedef __attribute__((ext_vector_type(4)))  unsigned int u32x4;

union FragU { bf16x16 v; u32x4 q[2]; };

// ---------- constants ----------
#define NB    32
#define CIN_  512
#define HH    8
#define WW    64
#define NCLS  6000
#define NCLSP 6016          // padded to multiple of 128
#define SMAX  25
#define LEXT  51            // 2*SMAX+1
#define K1    4608          // 512*9
#define PIXT  16384         // 32*8*64
#define NEGV  (-1e30f)
#define LOGTINY (-87.33654475055310898657f)

__device__ __forceinline__ unsigned short f2bf(float f) {
  unsigned int u = __float_as_uint(f);
  unsigned int r = u + 0x7FFFu + ((u >> 16) & 1u);   // RNE
  return (unsigned short)(r >> 16);
}
__device__ __forceinline__ float bflo(unsigned int u) {
  return __uint_as_float(u << 16);
}
__device__ __forceinline__ float bfhi(unsigned int u) {
  return __uint_as_float(u & 0xffff0000u);
}
__device__ __forceinline__ float logaddexpf_(float a, float b) {
  float m = fmaxf(a, b);
  float d = fminf(a, b) - m;
  return m + log1pf(expf(d));
}

// ---------- prep kernels ----------
__global__ void k_prep_img(const float* __restrict__ img, unsigned short* __restrict__ imgT) {
  int idx = blockIdx.x * blockDim.x + threadIdx.x;
  if (idx >= NB * CIN_ * HH * WW) return;
  int w = idx & 63, h = (idx >> 6) & 7, c = (idx >> 9) & 511, n = idx >> 18;
  imgT[(size_t)(((n * 8 + h) * 64 + w)) * 512 + c] = f2bf(img[idx]);
}

__global__ void k_prep_w1(const float* __restrict__ mw1, const float* __restrict__ cw1,
                          const float* __restrict__ mb1, const float* __restrict__ cb1,
                          unsigned short* __restrict__ Wr1, float* __restrict__ bias1) {
  int idx = blockIdx.x * blockDim.x + threadIdx.x;
  if (idx < 512) bias1[idx] = (idx < 256) ? mb1[idx] : cb1[idx - 256];
  if (idx >= 512 * K1) return;
  int co = idx / K1;
  int r  = idx - co * K1;
  int khw = r >> 9;          // 0..8 = kh*3+kw
  int ci  = r & 511;
  float v = (co < 256) ? mw1[((size_t)co * 512 + ci) * 9 + khw]
                       : cw1[((size_t)(co - 256) * 512 + ci) * 9 + khw];
  Wr1[idx] = f2bf(v);
}

__global__ void k_prep_cw2(const float* __restrict__ cw2, const float* __restrict__ cb2,
                           unsigned short* __restrict__ Wc2, float* __restrict__ cb2p) {
  int idx = blockIdx.x * blockDim.x + threadIdx.x;
  if (idx < NCLSP) cb2p[idx] = (idx < NCLS) ? cb2[idx] : NEGV;
  if (idx >= NCLSP * 256) return;
  int cls = idx >> 8, k = idx & 255;
  Wc2[idx] = (cls < NCLS) ? f2bf(cw2[(size_t)cls * 256 + k]) : (unsigned short)0;
}

// ---------- shared GEMM helpers ----------
__device__ __forceinline__ void g_wmma(const FragU* a, const FragU* b, f32x8 acc[4][2]) {
#pragma unroll
  for (int i = 0; i < 4; ++i)
#pragma unroll
    for (int j = 0; j < 2; ++j)
      acc[i][j] = __builtin_amdgcn_wmma_f32_16x16x32_bf16(
          false, a[i].v, false, b[j].v, (short)0, acc[i][j], false, false);
}

struct G1Ctx {
  const unsigned short* Wr1;
  const unsigned short* imgT;
  int coBase, lm, lh;
  int pn[2], py[2], px[2];
};

__device__ __forceinline__ void g1_load(const G1Ctx& c, int kk, FragU* a, FragU* b) {
  const int khw = kk >> 9;
  const int ci0 = kk & 511;
  const int kh = khw / 3, kw = khw - kh * 3;
#pragma unroll
  for (int i = 0; i < 4; ++i) {                 // A: lane M=l%16, K runs kb & kb+16
    int co = c.coBase + i * 16 + c.lm;
    const unsigned short* p = c.Wr1 + (size_t)co * K1 + kk + (c.lh << 3);
    a[i].q[0] = *(const u32x4*)p;
    a[i].q[1] = *(const u32x4*)(p + 16);
  }
#pragma unroll
  for (int j = 0; j < 2; ++j) {                 // B: lane N=l%16, K runs kb & kb+16
    int y = c.py[j] + kh - 1, x = c.px[j] + kw - 1;
    bool valid = ((unsigned)y < 8u) && ((unsigned)x < 64u);
    int yc = min(max(y, 0), 7), xc = min(max(x, 0), 63);   // clamped safe address
    const unsigned short* p =
        c.imgT + (size_t)(((c.pn[j] << 3) + yc) * 64 + xc) * 512 + ci0 + (c.lh << 3);
    u32x4 t0 = *(const u32x4*)p;
    u32x4 t1 = *(const u32x4*)(p + 16);
    u32x4 z = (u32x4)(0u);
    b[j].q[0] = valid ? t0 : z;                 // branch-free zero-padding
    b[j].q[1] = valid ? t1 : z;
  }
}

// ---------- GEMM1: implicit 3x3 conv, D[co=512][pix=16384], K=4608 ----------
__global__ __launch_bounds__(256) void k_gemm1(
    const unsigned short* __restrict__ Wr1,    // [512][4608] bf16, k = khw*512+ci
    const unsigned short* __restrict__ imgT,   // [pix][512]  bf16
    const float* __restrict__ bias1,           // [512]
    unsigned short* __restrict__ hidden) {     // [pix][512]  bf16
  const int lane = threadIdx.x & 31;
  const int wave = threadIdx.x >> 5;
  const int wm = wave >> 2, wn = wave & 3;     // 2 x 4 wave grid
  const int coBase  = blockIdx.x * 128 + wm * 64;
  const int pixBase = blockIdx.y * 128 + wn * 32;
  const int lm = lane & 15, lh = lane >> 4;

  G1Ctx c;
  c.Wr1 = Wr1; c.imgT = imgT; c.coBase = coBase; c.lm = lm; c.lh = lh;
#pragma unroll
  for (int j = 0; j < 2; ++j) {
    int pix = pixBase + j * 16 + lm;
    c.pn[j] = pix >> 9; int rem = pix & 511; c.py[j] = rem >> 6; c.px[j] = rem & 63;
  }

  f32x8 acc[4][2];
#pragma unroll
  for (int i = 0; i < 4; ++i)
#pragma unroll
    for (int j = 0; j < 2; ++j) acc[i][j] = (f32x8)(0.0f);

  // 2-stage software pipeline over 144 k-steps
  FragU A0[4], B0[2], A1[4], B1[2];
  g1_load(c, 0, A0, B0);
  int kk = 32;
#pragma unroll 1
  for (int it = 0; it < 71; ++it) {
    g1_load(c, kk, A1, B1);
    g_wmma(A0, B0, acc);
    g1_load(c, kk + 32, A0, B0);
    g_wmma(A1, B1, acc);
    kk += 64;
  }
  g1_load(c, K1 - 32, A1, B1);
  g_wmma(A0, B0, acc);
  g_wmma(A1, B1, acc);

  // epilogue: vector bias load + packed b128 stores (8 contiguous bf16 per lane)
#pragma unroll
  for (int i = 0; i < 4; ++i) {
    const float* bp = bias1 + coBase + i * 16 + (lh << 3);
    f32x4 bb0 = *(const f32x4*)bp;
    f32x4 bb1 = *(const f32x4*)(bp + 4);
    float barr[8];
#pragma unroll
    for (int t = 0; t < 4; ++t) { barr[t] = bb0[t]; barr[4 + t] = bb1[t]; }
#pragma unroll
    for (int j = 0; j < 2; ++j) {
      int pix = pixBase + j * 16 + lm;
      u32x4 st;
#pragma unroll
      for (int v = 0; v < 4; ++v) {
        unsigned lo = f2bf(acc[i][j][2 * v]     + barr[2 * v]);
        unsigned hi = f2bf(acc[i][j][2 * v + 1] + barr[2 * v + 1]);
        st[v] = lo | (hi << 16);
      }
      *(u32x4*)(hidden + (size_t)pix * 512 + coBase + i * 16 + (lh << 3)) = st;
    }
  }
}

// ---------- GEMM2: classes x pixels, fused online logsumexp over classes ----------
__device__ __forceinline__ void g2_load(
    const unsigned short* Wc2, const unsigned short* hidden,
    int ct, int wm, int pixBase, int lm, int lh, int kk, FragU* a, FragU* b) {
#pragma unroll
  for (int i = 0; i < 4; ++i) {
    int cls = ct + wm * 64 + i * 16 + lm;
    const unsigned short* p = Wc2 + (size_t)cls * 256 + kk + (lh << 3);
    a[i].q[0] = *(const u32x4*)p;
    a[i].q[1] = *(const u32x4*)(p + 16);
  }
#pragma unroll
  for (int j = 0; j < 2; ++j) {
    int pix = pixBase + j * 16 + lm;
    const unsigned short* p = hidden + (size_t)pix * 512 + 256 + kk + (lh << 3);
    b[j].q[0] = *(const u32x4*)p;
    b[j].q[1] = *(const u32x4*)(p + 16);
  }
}

__global__ __launch_bounds__(256) void k_gemm2_lse(
    const unsigned short* __restrict__ Wc2,     // [6016][256] bf16
    const unsigned short* __restrict__ hidden,  // [pix][512], classify at +256
    const float* __restrict__ cb2p,             // [6016], padded with -1e30
    float* __restrict__ pm, float* __restrict__ ps) {  // [2][pix] partial (max, sumexp)
  __shared__ float redm[2][128];
  __shared__ float reds[2][128];
  const int lane = threadIdx.x & 31;
  const int wave = threadIdx.x >> 5;
  const int wm = wave >> 2, wn = wave & 3;
  const int pixBase = blockIdx.x * 128 + wn * 32;
  const int lm = lane & 15, lh = lane >> 4;
  const int half = blockIdx.y;                  // split classes across 2 blocks
  const int ct0 = half ? 3072 : 0;
  const int ct1 = half ? NCLSP : 3072;

  float run_m[2] = {NEGV, NEGV};
  float run_s[2] = {0.f, 0.f};

  for (int ct = ct0; ct < ct1; ct += 128) {
    f32x8 acc[4][2];
#pragma unroll
    for (int i = 0; i < 4; ++i)
#pragma unroll
      for (int j = 0; j < 2; ++j) acc[i][j] = (f32x8)(0.0f);

    // pipelined K=256 (8 steps of 32)
    FragU A0[4], B0[2], A1[4], B1[2];
    g2_load(Wc2, hidden, ct, wm, pixBase, lm, lh, 0, A0, B0);
#pragma unroll 1
    for (int it = 0; it < 3; ++it) {
      g2_load(Wc2, hidden, ct, wm, pixBase, lm, lh, 32 + it * 64, A1, B1);
      g_wmma(A0, B0, acc);
      g2_load(Wc2, hidden, ct, wm, pixBase, lm, lh, 64 + it * 64, A0, B0);
      g_wmma(A1, B1, acc);
    }
    g2_load(Wc2, hidden, ct, wm, pixBase, lm, lh, 224, A1, B1);
    g_wmma(A0, B0, acc);
    g_wmma(A1, B1, acc);

    // bias via vector loads (classes contiguous per lane)
#pragma unroll
    for (int i = 0; i < 4; ++i) {
      const float* bp = cb2p + ct + wm * 64 + i * 16 + (lh << 3);
      f32x4 bb0 = *(const f32x4*)bp;
      f32x4 bb1 = *(const f32x4*)(bp + 4);
      float barr[8];
#pragma unroll
      for (int t = 0; t < 4; ++t) { barr[t] = bb0[t]; barr[4 + t] = bb1[t]; }
#pragma unroll
      for (int j = 0; j < 2; ++j)
#pragma unroll
        for (int v = 0; v < 8; ++v) acc[i][j][v] += barr[v];
    }

    // online logsumexp per pixel (classes live in M: regs + lane halves)
#pragma unroll
    for (int j = 0; j < 2; ++j) {
      float tmax = NEGV;
#pragma unroll
      for (int i = 0; i < 4; ++i)
#pragma unroll
        for (int v = 0; v < 8; ++v) tmax = fmaxf(tmax, acc[i][j][v]);
      tmax = fmaxf(tmax, __shfl_xor(tmax, 16, 32));
      float tsum = 0.f;
#pragma unroll
      for (int i = 0; i < 4; ++i)
#pragma unroll
        for (int v = 0; v < 8; ++v) tsum += expf(acc[i][j][v] - tmax);
      tsum += __shfl_xor(tsum, 16, 32);
      if (tmax > run_m[j]) {
        run_s[j] = run_s[j] * expf(run_m[j] - tmax) + tsum;
        run_m[j] = tmax;
      } else {
        run_s[j] += tsum * expf(tmax - run_m[j]);
      }
    }
  }

  if (lane < 16) {
#pragma unroll
    for (int j = 0; j < 2; ++j) {
      int p = wn * 32 + j * 16 + lane;
      redm[wm][p] = run_m[j];
      reds[wm][p] = run_s[j];
    }
  }
  __syncthreads();
  if (threadIdx.x < 128) {
    int p = threadIdx.x;
    float m0 = redm[0][p], m1 = redm[1][p];
    float m = fmaxf(m0, m1);
    float s = reds[0][p] * expf(m0 - m) + reds[1][p] * expf(m1 - m);
    int pix = blockIdx.x * 128 + p;
    pm[half * PIXT + pix] = m;
    ps[half * PIXT + pix] = s;
  }
}

__global__ void k_lse_fin(const float* __restrict__ pm, const float* __restrict__ ps,
                          float* __restrict__ lse) {
  int idx = blockIdx.x * blockDim.x + threadIdx.x;
  if (idx >= PIXT) return;
  float m0 = pm[idx], m1 = pm[PIXT + idx];
  float m = fmaxf(m0, m1);
  float s = ps[idx] * expf(m0 - m) + ps[PIXT + idx] * expf(m1 - m);
  lse[idx] = m + logf(s);
}

// ---------- mask branch 1x1 conv + log-softmax over H ----------
__global__ __launch_bounds__(512) void k_mask(
    const unsigned short* __restrict__ hidden, const float* __restrict__ mw2,
    const float* __restrict__ mb2, float* __restrict__ log_mask) {
  __shared__ float mv[512];
  __shared__ float lseH[64];
  int n = blockIdx.x;
  int h = threadIdx.x >> 6, w = threadIdx.x & 63;
  int pix = (n * 8 + h) * 64 + w;
  float acc = mb2[0];
  const u32x4* hp = (const u32x4*)(hidden + (size_t)pix * 512);
#pragma unroll 4
  for (int k8 = 0; k8 < 32; ++k8) {
    u32x4 hq = hp[k8];
    const float* wp = mw2 + k8 * 8;
#pragma unroll
    for (int cc = 0; cc < 4; ++cc) {
      unsigned u = hq[cc];
      acc += bflo(u) * wp[2 * cc] + bfhi(u) * wp[2 * cc + 1];
    }
  }
  mv[threadIdx.x] = acc;
  __syncthreads();
  if (threadIdx.x < 64) {
    int ww = threadIdx.x;
    float m = NEGV;
    for (int hh = 0; hh < 8; ++hh) m = fmaxf(m, mv[hh * 64 + ww]);
    float s = 0.f;
    for (int hh = 0; hh < 8; ++hh) s += expf(mv[hh * 64 + ww] - m);
    lseH[ww] = m + logf(s);
  }
  __syncthreads();
  log_mask[pix] = mv[threadIdx.x] - lseH[w];
}

// ---------- gather per-(n,label-slot) emit tensor E[n][26][h][w] ----------
__global__ __launch_bounds__(512) void k_gather(
    const unsigned short* __restrict__ hidden, const float* __restrict__ cw2,
    const float* __restrict__ cb2, const int* __restrict__ label,
    const float* __restrict__ log_mask, const float* __restrict__ lse,
    float* __restrict__ E) {
  int n = blockIdx.x / 26;
  int j = blockIdx.x % 26;
  int cls = (j == 0) ? 0 : label[n * SMAX + (j - 1)];
  int h = threadIdx.x >> 6, w = threadIdx.x & 63;
  int pix = (n * 8 + h) * 64 + w;
  float acc = cb2[cls];
  const u32x4* hp = (const u32x4*)(hidden + (size_t)pix * 512 + 256);
  const float* wrow = cw2 + (size_t)cls * 256;
#pragma unroll 4
  for (int k8 = 0; k8 < 32; ++k8) {
    u32x4 hq = hp[k8];
    const float* wp = wrow + k8 * 8;
#pragma unroll
    for (int cc = 0; cc < 4; ++cc) {
      unsigned u = hq[cc];
      acc += bflo(u) * wp[2 * cc] + bfhi(u) * wp[2 * cc + 1];
    }
  }
  float e = log_mask[pix] + acc - lse[pix];
  e = fmaxf(e, LOGTINY);
  E[(((size_t)n * 26 + j) * 8 + h) * 64 + w] = e;
}

// ---------- 2D CTC: one block per sample; alpha[h][s] scan over T=64 ----------
__global__ __launch_bounds__(512) void k_ctc(
    const float* __restrict__ E, const int* __restrict__ label,
    const int* __restrict__ length, float* __restrict__ out) {
  __shared__ float al[8][64];
  __shared__ float bb[64];
  int n = blockIdx.x;
  int h = threadIdx.x >> 6, s = threadIdx.x & 63;
  bool act = (s < LEXT);
  int jj = 0; bool skip = false;
  if (act && (s & 1)) {
    jj = (s + 1) >> 1;
    if (s >= 3)
      skip = (label[n * SMAX + ((s - 1) >> 1)] != label[n * SMAX + ((s - 3) >> 1)]);
  }
  const float* En = E + (size_t)n * 26 * 8 * 64;
  if (act) {
    float e0 = En[((size_t)jj * 8 + h) * 64 + 0];
    al[h][s] = (s < 2) ? e0 : NEGV;
  }
  for (int t = 1; t < WW; ++t) {
    __syncthreads();                            // alpha stable
    if (h == 0 && act) {                        // marginalize over height
      float m = NEGV;
#pragma unroll
      for (int hh = 0; hh < 8; ++hh) m = fmaxf(m, al[hh][s]);
      float ss = 0.f;
#pragma unroll
      for (int hh = 0; hh < 8; ++hh) ss += expf(al[hh][s] - m);
      bb[s] = m + logf(ss);
    }
    __syncthreads();                            // bb visible
    if (act) {
      float tr = bb[s];
      if (s >= 1) tr = logaddexpf_(tr, bb[s - 1]);
      if (skip)   tr = logaddexpf_(tr, bb[s - 2]);
      al[h][s] = En[((size_t)jj * 8 + h) * 64 + t] + tr;
    }
  }
  __syncthreads();
  if (threadIdx.x == 0) {
    int len = length[n];
    int last = 2 * len;
    float f[2];
    for (int q = 0; q < 2; ++q) {
      int si = last - q;
      float m = NEGV;
      for (int hh = 0; hh < 8; ++hh) m = fmaxf(m, al[hh][si]);
      float ss = 0.f;
      for (int hh = 0; hh < 8; ++hh) ss += expf(al[hh][si] - m);
      f[q] = m + logf(ss);
    }
    float loss = -logaddexpf_(f[0], f[1]) / (float)len;
    if (isinf(loss)) loss = 6.9f;
    out[n] = loss;
  }
}

// ---------- launch ----------
extern "C" void kernel_launch(void* const* d_in, const int* in_sizes, int n_in,
                              void* d_out, int out_size, void* d_ws, size_t ws_size,
                              hipStream_t stream) {
  const float* img   = (const float*)d_in[0];
  const int*   label = (const int*)d_in[1];
  const int*   length= (const int*)d_in[2];
  const float* mw1   = (const float*)d_in[3];
  const float* mb1   = (const float*)d_in[4];
  const float* mw2   = (const float*)d_in[5];
  const float* mb2   = (const float*)d_in[6];
  const float* cw1   = (const float*)d_in[7];
  const float* cb1   = (const float*)d_in[8];
  const float* cw2   = (const float*)d_in[9];
  const float* cb2   = (const float*)d_in[10];
  float* out = (float*)d_out;

  char* ws = (char*)d_ws;
  const size_t o_imgT   = 0;                         // 16,777,216 B
  const size_t o_Wr1    = o_imgT   + 16777216;       //  4,718,592 B
  const size_t o_Wc2    = o_Wr1    + 4718592;        //  3,080,192 B
  const size_t o_hidden = o_Wc2    + 3080192;        // 16,777,216 B
  const size_t o_bias1  = o_hidden + 16777216;       //      2,048 B
  const size_t o_cb2p   = o_bias1  + 2048;           //     24,064 B
  const size_t o_lse    = o_cb2p   + 24064;          //     65,536 B
  const size_t o_lm     = o_lse    + 65536;          //     65,536 B
  const size_t o_E      = o_lm     + 65536;          //  1,703,936 B
  const size_t o_pm     = o_E      + 1703936;        //    131,072 B
  const size_t o_ps     = o_pm     + 131072;         //    131,072 B

  unsigned short* imgT   = (unsigned short*)(ws + o_imgT);
  unsigned short* Wr1    = (unsigned short*)(ws + o_Wr1);
  unsigned short* Wc2    = (unsigned short*)(ws + o_Wc2);
  unsigned short* hidden = (unsigned short*)(ws + o_hidden);
  float* bias1   = (float*)(ws + o_bias1);
  float* cb2p    = (float*)(ws + o_cb2p);
  float* lse     = (float*)(ws + o_lse);
  float* log_mask= (float*)(ws + o_lm);
  float* E       = (float*)(ws + o_E);
  float* pm      = (float*)(ws + o_pm);
  float* ps      = (float*)(ws + o_ps);

  k_prep_img<<<(NB * CIN_ * HH * WW + 255) / 256, 256, 0, stream>>>(img, imgT);
  k_prep_w1<<<(512 * K1 + 255) / 256, 256, 0, stream>>>(mw1, cw1, mb1, cb1, Wr1, bias1);
  k_prep_cw2<<<(NCLSP * 256 + 255) / 256, 256, 0, stream>>>(cw2, cb2, Wc2, cb2p);

  k_gemm1<<<dim3(4, 128), 256, 0, stream>>>(Wr1, imgT, bias1, hidden);
  k_gemm2_lse<<<dim3(PIXT / 128, 2), 256, 0, stream>>>(Wc2, hidden, cb2p, pm, ps);
  k_lse_fin<<<(PIXT + 255) / 256, 256, 0, stream>>>(pm, ps, lse);
  k_mask<<<NB, 512, 0, stream>>>(hidden, mw2, mb2, log_mask);
  k_gather<<<NB * 26, 512, 0, stream>>>(hidden, cw2, cb2, label, log_mask, lse, E);
  k_ctc<<<NB, 512, 0, stream>>>(E, label, length, out);
}
